// GAT_11665131176405
// MI455X (gfx1250) — matrix operand order repaired and checked by reading, not compiled, not run on hardware
//
#include <hip/hip_runtime.h>
#include <hip/hip_bf16.h>
#include <stdint.h>

// ---------- types for WMMA (gfx1250, wave32) ----------
typedef __attribute__((ext_vector_type(16))) __bf16   v16bf;
typedef __attribute__((ext_vector_type(8)))  float    v8f;
typedef __attribute__((ext_vector_type(4)))  unsigned v4u;

union FragBF { v4u q[2]; v16bf v; };   // ext-vector members -> trivially copyable union

// ---------- order-preserving float<->uint for atomicMax segment-max ----------
__device__ __forceinline__ unsigned ordEnc(float f) {
  unsigned u = __float_as_uint(f);
  return (u & 0x80000000u) ? ~u : (u | 0x80000000u);
}
__device__ __forceinline__ float ordDec(unsigned u) {
  unsigned b = (u & 0x80000000u) ? (u & 0x7FFFFFFFu) : ~u;
  return __uint_as_float(b);
}

// ---------- utility kernels ----------
__global__ void k_zero(unsigned* p, int n) {
  int i = blockIdx.x * blockDim.x + threadIdx.x;
  if (i < n) p[i] = 0u;
}

__global__ void k_f32_to_bf16(const float* __restrict__ in, __bf16* __restrict__ out, int n) {
  int i = blockIdx.x * blockDim.x + threadIdx.x;
  if (i < n) out[i] = (__bf16)in[i];
}

// Wt[n][k] = W[k][n] as bf16; zero-pad rows Nvalid..Npad-1 (keeps WMMA B loads valid)
__global__ void k_transpose_w(const float* __restrict__ W, __bf16* __restrict__ Wt,
                              int K, int Nvalid, int Npad) {
  int i = blockIdx.x * blockDim.x + threadIdx.x;
  if (i >= K * Npad) return;
  int n = i / K, k = i - n * K;
  Wt[(size_t)n * K + k] = (n < Nvalid) ? (__bf16)W[(size_t)k * Nvalid + n] : (__bf16)0.0f;
}

__global__ void k_reduce_sum(const float* __restrict__ x, int n, float* out) {
  float s = 0.f;
  for (int i = blockIdx.x * blockDim.x + threadIdx.x; i < n; i += gridDim.x * blockDim.x)
    s += x[i];
  atomicAdd(out, s);
}

// ---------- WMMA GEMM: C[M x ldC] = A[M x K]*(Wt[Npad x K])^T + bias ----------
// One wave per 16x16 tile; 4 waves per block along M. K % 32 == 0. M % 16 == 0.
__global__ __launch_bounds__(128) void k_gemm_wmma(
    const __bf16* __restrict__ A, const __bf16* __restrict__ Wt,
    const float* __restrict__ bias, float* __restrict__ C,
    int M, int K, int ldC, int Nvalid) {
  const int wave = threadIdx.x >> 5;
  const int lane = threadIdx.x & 31;
  const int tm   = blockIdx.x * 4 + wave;     // tile row (wave-uniform branch below)
  if (tm * 16 >= M) return;
  const int tn  = blockIdx.y;
  const int l15 = lane & 15;
  const int hi  = lane >> 4;                  // lane group 0/1
  const int n   = tn * 16 + l15;

  const __bf16* Arow = A  + (size_t)(tm * 16 + l15) * K;   // A fragment: row m
  const __bf16* Bcol = Wt + (size_t)n * K + hi * 8;         // B fragment: column n (pre-transposed)

  v8f acc;
#pragma unroll
  for (int j = 0; j < 8; ++j) acc[j] = 0.f;

  for (int k0 = 0; k0 < K; k0 += 32) {
    FragBF a, b;
    // 16-bit A 16x32 layout: lanes0-15 K={k0..k0+7, k0+16..k0+23}, lanes16-31 shift by 8
    a.q[0] = *(const v4u*)(Arow + k0 + hi * 8);
    a.q[1] = *(const v4u*)(Arow + k0 + 16 + hi * 8);
    // B 32x16 layout: lane l15 = column n; lanes0-15 K={k0..k0+7, k0+16..k0+23}, lanes16-31 shift by 8
    b.q[0] = *(const v4u*)(Bcol + k0);
    b.q[1] = *(const v4u*)(Bcol + k0 + 16);
    if (k0 + 32 < K) __builtin_prefetch(Arow + k0 + 32, 0, 1);   // global_prefetch_b8
    acc = __builtin_amdgcn_wmma_f32_16x16x32_bf16(
        /*neg_a=*/false, a.v, /*neg_b=*/false, b.v,
        /*c_mod=*/(short)0, acc, /*reuse_a=*/false, /*reuse_b=*/false);
  }

  if (n < Nvalid) {
    float bn = bias ? bias[n] : 0.f;
#pragma unroll
    for (int j = 0; j < 8; ++j) {  // C/D layout: VGPR j -> row hi*8+j, col l15
      C[(size_t)(tm * 16 + hi * 8 + j) * ldC + n] = acc[j] + bn;
    }
  }
}

// ---------- edge kernels ----------
// logits[e,h] = sum_c att[h,c] * leaky_relu(xl[src,h,c] + xr[dst,h,c] + ea*We[h,c]); fused segment-max
__global__ void k_edge_logits(const float* __restrict__ xl, const float* __restrict__ xr, int ldx,
                              const int* __restrict__ src, const int* __restrict__ dst,
                              const float* __restrict__ ea, const float* __restrict__ easum, float invE,
                              const float* __restrict__ We, const float* __restrict__ att,
                              float* __restrict__ logits, unsigned* __restrict__ nmax,
                              int E, int ET, int H, int Cc) {
  int t = blockIdx.x * blockDim.x + threadIdx.x;
  if (t >= ET * H) return;
  int e = t / H, h = t - e * H;
  int s = (e < E) ? src[e] : (e - E);
  int d = (e < E) ? dst[e] : (e - E);
  float eav = (e < E) ? ea[e] : (easum[0] * invE);   // self-loop edge_attr = mean
  const float* pl = xl + (size_t)s * ldx + h * Cc;
  const float* pr = xr + (size_t)d * ldx + h * Cc;
  const float* pw = We + h * Cc;
  const float* pa = att + h * Cc;
  float a = 0.f;
  for (int c = 0; c < Cc; ++c) {
    float m = pl[c] + pr[c] + eav * pw[c];
    m = (m > 0.f) ? m : 0.2f * m;
    a += pa[c] * m;
  }
  logits[t] = a;
  atomicMax(&nmax[(size_t)d * H + h], ordEnc(a));
}

// a = exp(logit - max); in-place; fused segment-sum
__global__ void k_edge_expsum(float* logits, const int* __restrict__ dst,
                              const unsigned* __restrict__ nmax, float* __restrict__ nsum,
                              int E, int ET, int H) {
  int t = blockIdx.x * blockDim.x + threadIdx.x;
  if (t >= ET * H) return;
  int e = t / H, h = t - e * H;
  int d = (e < E) ? dst[e] : (e - E);
  float a = __expf(logits[t] - ordDec(nmax[(size_t)d * H + h]));
  logits[t] = a;
  atomicAdd(&nsum[(size_t)d * H + h], a);
}

// hacc[dst,c] += (1/H) * sum_h (a[e,h]/nsum[dst,h]) * xl[src,h,c]   (head-mean fused)
__global__ void k_edge_scatter(const float* __restrict__ aexp, const float* __restrict__ nsum,
                               const float* __restrict__ xl, int ldx,
                               const int* __restrict__ src, const int* __restrict__ dst,
                               float* __restrict__ hacc, int E, int ET, int H, int Cc, float invH) {
  int t = blockIdx.x * blockDim.x + threadIdx.x;
  if (t >= ET * Cc) return;
  int e = t / Cc, c = t - e * Cc;
  int s = (e < E) ? src[e] : (e - E);
  int d = (e < E) ? dst[e] : (e - E);
  float a = 0.f;
  for (int h = 0; h < H; ++h)
    a += (aexp[(size_t)e * H + h] / nsum[(size_t)d * H + h]) * xl[(size_t)s * ldx + h * Cc + c];
  atomicAdd(&hacc[(size_t)d * Cc + c], a * invH);
}

// out = [relu](hacc + bias); optional f32 and/or bf16 destinations
__global__ void k_finalize(const float* __restrict__ hacc, const float* __restrict__ bias,
                           int Cc, int total, int doRelu,
                           float* __restrict__ fdst, __bf16* __restrict__ bdst) {
  int t = blockIdx.x * blockDim.x + threadIdx.x;
  if (t >= total) return;
  int c = t % Cc;
  float v = hacc[t] + bias[c];
  if (doRelu) v = fmaxf(v, 0.f);
  if (fdst) fdst[t] = v;
  if (bdst) bdst[t] = (__bf16)v;
}

// ---------- host orchestration ----------
extern "C" void kernel_launch(void* const* d_in, const int* in_sizes, int n_in,
                              void* d_out, int out_size, void* d_ws, size_t ws_size,
                              hipStream_t stream) {
  const float* x   = (const float*)d_in[0];
  const int*   ei  = (const int*)  d_in[1];
  const float* ea  = (const float*)d_in[2];
  const float* Wl1 = (const float*)d_in[3];  const float* bl1 = (const float*)d_in[4];
  const float* Wr1 = (const float*)d_in[5];  const float* br1 = (const float*)d_in[6];
  const float* We1 = (const float*)d_in[7];  const float* att1= (const float*)d_in[8];
  const float* b1  = (const float*)d_in[9];
  const float* Wl2 = (const float*)d_in[10]; const float* bl2 = (const float*)d_in[11];
  const float* Wr2 = (const float*)d_in[12]; const float* br2 = (const float*)d_in[13];
  const float* We2 = (const float*)d_in[14]; const float* att2= (const float*)d_in[15];
  const float* b2  = (const float*)d_in[16];
  const float* Wl3 = (const float*)d_in[17]; const float* bl3 = (const float*)d_in[18];
  const float* Wr3 = (const float*)d_in[19]; const float* br3 = (const float*)d_in[20];
  const float* We3 = (const float*)d_in[21]; const float* att3= (const float*)d_in[22];
  const float* b3  = (const float*)d_in[23];

  const int FIN = 512, H = 4, D = 32, L = 32, NC = 10;
  const int N  = in_sizes[0] / FIN;   // 50000
  const int E  = in_sizes[2];         // 800000
  const int ET = E + N;               // with self-loops
  const int* srcI = ei;
  const int* dstI = ei + E;

  // bump allocator over d_ws
  char* base = (char*)d_ws;
  size_t off = 0;
  auto alloc = [&](size_t bytes) -> void* {
    void* p = base + off;
    off += (bytes + 255) & ~(size_t)255;
    return p;
  };
  __bf16*  xbf   = (__bf16*) alloc((size_t)N * FIN * 2);
  __bf16*  wtl   = (__bf16*) alloc((size_t)128 * 512 * 2);
  __bf16*  wtr   = (__bf16*) alloc((size_t)128 * 512 * 2);
  float*   xl    = (float*)  alloc((size_t)N * 128 * 4);
  float*   xr    = (float*)  alloc((size_t)N * 128 * 4);
  float*   lg    = (float*)  alloc((size_t)ET * H * 4);
  unsigned* nmax = (unsigned*)alloc((size_t)N * H * 4);
  float*   nsum  = (float*)  alloc((size_t)N * H * 4);
  float*   hacc  = (float*)  alloc((size_t)N * 32 * 4);
  __bf16*  hbf   = (__bf16*) alloc((size_t)N * 32 * 2);
  float*   easum = (float*)  alloc(256);

  float* yout   = (float*)d_out;                    // N*NC
  float* latent = (float*)d_out + (size_t)N * NC;   // N*L

  auto cdiv = [](long long a, long long b) { return (int)((a + b - 1) / b); };
  const int T = 256;

  // edge_attr mean (self-loop fill value)
  k_zero<<<1, 64, 0, stream>>>((unsigned*)easum, 1);
  k_reduce_sum<<<256, T, 0, stream>>>(ea, E, easum);

  // ===== layer 1: 512 -> H*D (WMMA GEMMs) =====
  k_f32_to_bf16<<<cdiv((long long)N * FIN, T), T, 0, stream>>>(x, xbf, N * FIN);
  k_transpose_w<<<cdiv(512 * 128, T), T, 0, stream>>>(Wl1, wtl, 512, 128, 128);
  k_transpose_w<<<cdiv(512 * 128, T), T, 0, stream>>>(Wr1, wtr, 512, 128, 128);
  dim3 g1(cdiv(cdiv(N, 16), 4), 128 / 16);
  k_gemm_wmma<<<g1, 128, 0, stream>>>(xbf, wtl, bl1, xl, N, 512, 128, 128);
  k_gemm_wmma<<<g1, 128, 0, stream>>>(xbf, wtr, br1, xr, N, 512, 128, 128);
  k_zero<<<cdiv(N * H, T), T, 0, stream>>>(nmax, N * H);
  k_zero<<<cdiv(N * H, T), T, 0, stream>>>((unsigned*)nsum, N * H);
  k_zero<<<cdiv(N * D, T), T, 0, stream>>>((unsigned*)hacc, N * D);
  k_edge_logits<<<cdiv((long long)ET * H, T), T, 0, stream>>>(xl, xr, 128, srcI, dstI,
      ea, easum, 1.f / E, We1, att1, lg, nmax, E, ET, H, D);
  k_edge_expsum<<<cdiv((long long)ET * H, T), T, 0, stream>>>(lg, dstI, nmax, nsum, E, ET, H);
  k_edge_scatter<<<cdiv((long long)ET * D, T), T, 0, stream>>>(lg, nsum, xl, 128,
      srcI, dstI, hacc, E, ET, H, D, 1.f / H);
  k_finalize<<<cdiv(N * D, T), T, 0, stream>>>(hacc, b1, D, N * D, 1, (float*)nullptr, hbf);

  // ===== layer 2: D -> H*L =====
  k_transpose_w<<<cdiv(32 * 128, T), T, 0, stream>>>(Wl2, wtl, 32, 128, 128);
  k_transpose_w<<<cdiv(32 * 128, T), T, 0, stream>>>(Wr2, wtr, 32, 128, 128);
  k_gemm_wmma<<<g1, 128, 0, stream>>>(hbf, wtl, bl2, xl, N, 32, 128, 128);
  k_gemm_wmma<<<g1, 128, 0, stream>>>(hbf, wtr, br2, xr, N, 32, 128, 128);
  k_zero<<<cdiv(N * H, T), T, 0, stream>>>(nmax, N * H);
  k_zero<<<cdiv(N * H, T), T, 0, stream>>>((unsigned*)nsum, N * H);
  k_zero<<<cdiv(N * L, T), T, 0, stream>>>((unsigned*)hacc, N * L);
  k_edge_logits<<<cdiv((long long)ET * H, T), T, 0, stream>>>(xl, xr, 128, srcI, dstI,
      ea, easum, 1.f / E, We2, att2, lg, nmax, E, ET, H, L);
  k_edge_expsum<<<cdiv((long long)ET * H, T), T, 0, stream>>>(lg, dstI, nmax, nsum, E, ET, H);
  k_edge_scatter<<<cdiv((long long)ET * L, T), T, 0, stream>>>(lg, nsum, xl, 128,
      srcI, dstI, hacc, E, ET, H, L, 1.f / H);
  // latent = relu(h + b2) -> d_out (f32) and bf16 for layer 3
  k_finalize<<<cdiv(N * L, T), T, 0, stream>>>(hacc, b2, L, N * L, 1, latent, hbf);

  // ===== layer 3: L -> NC, H=1 (N padded to 16 in GEMM, ldC = 16) =====
  k_transpose_w<<<cdiv(32 * 16, T), T, 0, stream>>>(Wl3, wtl, 32, 10, 16);
  k_transpose_w<<<cdiv(32 * 16, T), T, 0, stream>>>(Wr3, wtr, 32, 10, 16);
  dim3 g3(cdiv(cdiv(N, 16), 4), 1);
  k_gemm_wmma<<<g3, 128, 0, stream>>>(hbf, wtl, bl3, xl, N, 32, 16, 10);
  k_gemm_wmma<<<g3, 128, 0, stream>>>(hbf, wtr, br3, xr, N, 32, 16, 10);
  k_zero<<<cdiv(N, T), T, 0, stream>>>(nmax, N);
  k_zero<<<cdiv(N, T), T, 0, stream>>>((unsigned*)nsum, N);
  k_zero<<<cdiv(N * NC, T), T, 0, stream>>>((unsigned*)hacc, N * NC);
  k_edge_logits<<<cdiv((long long)ET, T), T, 0, stream>>>(xl, xr, 16, srcI, dstI,
      ea, easum, 1.f / E, We3, att3, lg, nmax, E, ET, 1, NC);
  k_edge_expsum<<<cdiv((long long)ET, T), T, 0, stream>>>(lg, dstI, nmax, nsum, E, ET, 1);
  k_edge_scatter<<<cdiv((long long)ET * NC, T), T, 0, stream>>>(lg, nsum, xl, 16,
      srcI, dstI, hacc, E, ET, 1, NC, 1.f);
  // y = h + b3 (no relu) -> d_out
  k_finalize<<<cdiv(N * NC, T), T, 0, stream>>>(hacc, b3, NC, N * NC, 0, yout, (__bf16*)nullptr);

  (void)n_in; (void)out_size; (void)ws_size;
}
